// GAT_11751030522722
// MI455X (gfx1250) — matrix operand order repaired and checked by reading, not compiled, not run on hardware
//
#include <hip/hip_runtime.h>
#include <hip/hip_bf16.h>
#include <math.h>

// ---------------------------------------------------------------------------
// GAT (3-layer, DGL-style) for MI455X / gfx1250, wave32 + WMMA bf16.
// ---------------------------------------------------------------------------

#define NNODES   100000
#define NEDGES   1600000
#define NHEADS   8
#define NEG_SLOPE 0.2f
#define NCLS     41

typedef __attribute__((ext_vector_type(16))) __bf16 v16bf;
typedef __attribute__((ext_vector_type(8)))  float  v8f;

// ---------------- order-preserving float <-> uint key (for atomic max) -----
__device__ __forceinline__ unsigned enc_f32(float f) {
    unsigned b = __float_as_uint(f);
    return (b & 0x80000000u) ? ~b : (b | 0x80000000u);
}
__device__ __forceinline__ float dec_u32(unsigned k) {
    unsigned b = (k & 0x80000000u) ? (k & 0x7FFFFFFFu) : ~k;
    return __uint_as_float(b);
}
// encoded(-inf) == ~0xFF800000 == 0x007FFFFF
#define ENC_NEG_INF 0x007FFFFFu

__device__ __forceinline__ void pack16(v16bf& v, float4 a, float4 b,
                                       float4 c, float4 d)
{
    v[0]  = (__bf16)a.x; v[1]  = (__bf16)a.y; v[2]  = (__bf16)a.z; v[3]  = (__bf16)a.w;
    v[4]  = (__bf16)b.x; v[5]  = (__bf16)b.y; v[6]  = (__bf16)b.z; v[7]  = (__bf16)b.w;
    v[8]  = (__bf16)c.x; v[9]  = (__bf16)c.y; v[10] = (__bf16)c.z; v[11] = (__bf16)c.w;
    v[12] = (__bf16)d.x; v[13] = (__bf16)d.y; v[14] = (__bf16)d.z; v[15] = (__bf16)d.w;
}

// ---------------------------------------------------------------------------
// Kernel 0: transpose + zero-pad W [K,Nd] -> Wt [Ndpad,K]
// ---------------------------------------------------------------------------
__global__ void gat_transpose_pad(const float* __restrict__ W,
                                  float* __restrict__ Wt,
                                  int K, int Nd, int Ndpad)
{
    int idx = blockIdx.x * blockDim.x + threadIdx.x;   // c*K + k
    if (idx >= Ndpad * K) return;
    int c = idx / K, k = idx - c * K;
    Wt[idx] = (c < Nd) ? W[(size_t)k * Nd + c] : 0.0f;
}

// ---------------------------------------------------------------------------
// Kernel 1: C = A @ Bt^T  (A: [M,K] f32, Bt: [Ndpad,K] f32, C: [M,Nd] f32)
// One wave per 16x32 output tile (two 16x16 WMMA accumulators sharing A).
// Requirements: M % 16 == 0, K % 32 == 0, Ndpad % 32 == 0 (zero-padded).
// All hot-loop loads are unguarded b128; only the store checks col < Nd.
// ---------------------------------------------------------------------------
__global__ void gat_gemm_wmma(const float* __restrict__ A,
                              const float* __restrict__ Bt,
                              float* __restrict__ C,
                              int K, int Nd)
{
    const int lane = threadIdx.x;            // 0..31 (wave32)
    const int row0 = blockIdx.x * 16;        // M tile
    const int col0 = blockIdx.y * 32;        // pair of N tiles
    const int l15  = lane & 15;
    const int a_kb = (lane >> 4) * 16;       // 0 or 16
    const int b_kb = (lane >> 4) * 16;       // 0 or 16

    const float* __restrict__ ap  = A  + (size_t)(row0 + l15) * K;
    const float* __restrict__ bp0 = Bt + (size_t)(col0 + l15) * K;
    const float* __restrict__ bp1 = Bt + (size_t)(col0 + 16 + l15) * K;

    v8f acc0 = {}, acc1 = {};
    for (int k0 = 0; k0 < K; k0 += 32) {
        // A fragment: contiguous K = kb+0..15 (matches B)
        const float4* apv = (const float4*)(ap + k0 + a_kb);
        v16bf a;
        pack16(a, apv[0], apv[1], apv[2], apv[3]);

        // B fragments: contiguous K = kb+0..15 (matches A); Bt rows are contiguous in K
        const float4* b0v = (const float4*)(bp0 + k0 + b_kb);
        const float4* b1v = (const float4*)(bp1 + k0 + b_kb);
        v16bf b0, b1;
        pack16(b0, b0v[0], b0v[1], b0v[2], b0v[3]);
        pack16(b1, b1v[0], b1v[1], b1v[2], b1v[3]);

        acc0 = __builtin_amdgcn_wmma_f32_16x16x32_bf16(
                   false, a, false, b0, (short)0, acc0, false, false);
        acc1 = __builtin_amdgcn_wmma_f32_16x16x32_bf16(
                   false, a, false, b1, (short)0, acc1, false, false);
    }

    // C/D layout: row = 8*(lane>>4)+i, col = lane&15
    const int c0 = col0 + l15;
    const int c1 = col0 + 16 + l15;
    const int r0 = row0 + (lane >> 4) * 8;
#pragma unroll
    for (int i = 0; i < 8; ++i) {
        float* crow = C + (size_t)(r0 + i) * Nd;
        if (c0 < Nd) crow[c0] = acc0[i];
        if (c1 < Nd) crow[c1] = acc1[i];
    }
}

// ---------------------------------------------------------------------------
// Kernel 2: el/er attention terms.  idx = n*H + h
// ---------------------------------------------------------------------------
__global__ void gat_attn_terms(const float* __restrict__ z,
                               const float* __restrict__ al,
                               const float* __restrict__ ar,
                               float* __restrict__ el,
                               float* __restrict__ er,
                               int F)
{
    int idx = blockIdx.x * blockDim.x + threadIdx.x;
    if (idx >= NNODES * NHEADS) return;
    int n = idx >> 3, h = idx & 7;
    const float* zr  = z  + ((size_t)n * NHEADS + h) * F;
    const float* alr = al + h * F;
    const float* arr = ar + h * F;
    float sl = 0.f, sr = 0.f;
    for (int f = 0; f < F; ++f) {
        float v = zr[f];
        sl += v * alr[f];
        sr += v * arr[f];
    }
    el[idx] = sl;
    er[idx] = sr;
}

// ---------------------------------------------------------------------------
// Kernel 3: init segment-max keys
// ---------------------------------------------------------------------------
__global__ void gat_fill_u32(unsigned* __restrict__ p, unsigned v, int n)
{
    int i = blockIdx.x * blockDim.x + threadIdx.x;
    if (i < n) p[i] = v;
}

// ---------------------------------------------------------------------------
// Kernel 4: edge score = leaky_relu(el[src]+er[dst]); segment max via
// atomicMax on monotone uint keys.  idx = e*H + h
// ---------------------------------------------------------------------------
__global__ void gat_edge_score_max(const float* __restrict__ el,
                                   const float* __restrict__ er,
                                   const int* __restrict__ src,
                                   const int* __restrict__ dst,
                                   float* __restrict__ ebuf,
                                   unsigned* __restrict__ menc)
{
    int idx = blockIdx.x * blockDim.x + threadIdx.x;
    if (idx >= NEDGES * NHEADS) return;
    int e = idx >> 3, h = idx & 7;
    int s = src[e], d = dst[e];
    float v = el[s * NHEADS + h] + er[d * NHEADS + h];
    v = (v > 0.f) ? v : NEG_SLOPE * v;
    ebuf[idx] = v;
    atomicMax(&menc[d * NHEADS + h], enc_f32(v));
}

// ---------------------------------------------------------------------------
// Kernel 5: ex = exp(e - m[dst]); segment sum of ex.  idx = e*H + h
// ---------------------------------------------------------------------------
__global__ void gat_edge_exp_sum(float* __restrict__ ebuf,
                                 const unsigned* __restrict__ menc,
                                 float* __restrict__ sden,
                                 const int* __restrict__ dst)
{
    int idx = blockIdx.x * blockDim.x + threadIdx.x;
    if (idx >= NEDGES * NHEADS) return;
    int e = idx >> 3, h = idx & 7;
    int d = dst[e];
    float m = dec_u32(menc[d * NHEADS + h]);
    // nodes with no in-edges keep m == -inf -> treat as 0 (reference semantics)
    if ((__float_as_uint(m) & 0x7F800000u) == 0x7F800000u) m = 0.f;
    float ex = __expf(ebuf[idx] - m);
    ebuf[idx] = ex;
    atomicAdd(&sden[d * NHEADS + h], ex);
}

// ---------------------------------------------------------------------------
// Kernel 6: aggregation. One block per edge, float4 lanes over H*F features.
// agg[dst] += alpha * z[src]
// ---------------------------------------------------------------------------
__global__ void gat_aggregate(const float* __restrict__ z,
                              const float* __restrict__ ebuf,
                              const float* __restrict__ sden,
                              const int* __restrict__ src,
                              const int* __restrict__ dst,
                              float* __restrict__ agg,
                              int Dout, int F)
{
    int e = blockIdx.x;
    int j = threadIdx.x;                 // float4 index
    if (j >= (Dout >> 2)) return;
    int s = src[e], d = dst[e];
    int h = (4 * j) / F;
    float den = sden[d * NHEADS + h];
    float alpha = ebuf[(size_t)e * NHEADS + h] / fmaxf(den, 1e-12f);
    const float4* zrow = (const float4*)(z + (size_t)s * Dout);
    float4 zv = zrow[j];
    float* ag = agg + (size_t)d * Dout + 4 * j;
    atomicAdd(ag + 0, alpha * zv.x);
    atomicAdd(ag + 1, alpha * zv.y);
    atomicAdd(ag + 2, alpha * zv.z);
    atomicAdd(ag + 3, alpha * zv.w);
}

// ---------------------------------------------------------------------------
// Kernel 7: hidden finalize: hout[n,f] = mean_h relu(agg[n,h,f] + b[h,f])
// ---------------------------------------------------------------------------
__global__ void gat_finalize_hidden(const float* __restrict__ agg,
                                    const float* __restrict__ bias,
                                    float* __restrict__ hout,
                                    int F)
{
    int idx = blockIdx.x * blockDim.x + threadIdx.x;
    if (idx >= NNODES * F) return;
    int n = idx / F, f = idx - n * F;
    float acc = 0.f;
#pragma unroll
    for (int h = 0; h < NHEADS; ++h) {
        float v = agg[((size_t)n * NHEADS + h) * F + f] + bias[h * F + f];
        acc += (v > 0.f) ? v : 0.f;
    }
    hout[idx] = acc * (1.0f / NHEADS);
}

// ---------------------------------------------------------------------------
// Kernel 8: final layer: mean over heads + softmax over classes
// ---------------------------------------------------------------------------
__global__ void gat_finalize_softmax(const float* __restrict__ agg,
                                     const float* __restrict__ bias,
                                     float* __restrict__ out)
{
    int n = blockIdx.x * blockDim.x + threadIdx.x;
    if (n >= NNODES) return;
    float logits[NCLS];
    float mx = -INFINITY;
#pragma unroll 1
    for (int c = 0; c < NCLS; ++c) {
        float s = 0.f;
#pragma unroll
        for (int h = 0; h < NHEADS; ++h)
            s += agg[((size_t)n * NHEADS + h) * NCLS + c] + bias[h * NCLS + c];
        s *= (1.0f / NHEADS);
        logits[c] = s;
        mx = fmaxf(mx, s);
    }
    float sum = 0.f;
#pragma unroll 1
    for (int c = 0; c < NCLS; ++c) {
        float ex = __expf(logits[c] - mx);
        logits[c] = ex;
        sum += ex;
    }
    float inv = 1.0f / sum;
#pragma unroll 1
    for (int c = 0; c < NCLS; ++c)
        out[(size_t)n * NCLS + c] = logits[c] * inv;
}

// ---------------------------------------------------------------------------
// Host launcher
// ---------------------------------------------------------------------------
extern "C" void kernel_launch(void* const* d_in, const int* in_sizes, int n_in,
                              void* d_out, int out_size, void* d_ws, size_t ws_size,
                              hipStream_t stream)
{
    (void)in_sizes; (void)n_in; (void)out_size; (void)ws_size;

    const float* x   = (const float*)d_in[0];
    const int*   src = (const int*)  d_in[1];
    const int*   dst = (const int*)  d_in[2];
    const float* W [3] = { (const float*)d_in[3],  (const float*)d_in[7],  (const float*)d_in[11] };
    const float* al[3] = { (const float*)d_in[4],  (const float*)d_in[8],  (const float*)d_in[12] };
    const float* ar[3] = { (const float*)d_in[5],  (const float*)d_in[9],  (const float*)d_in[13] };
    const float* bb[3] = { (const float*)d_in[6],  (const float*)d_in[10], (const float*)d_in[14] };

    // Workspace carve-up (256B aligned)
    char* ws = (char*)d_ws;
    size_t off = 0;
    auto carve = [&](size_t bytes) -> void* {
        void* p = ws + off;
        off += (bytes + 255) & ~(size_t)255;
        return p;
    };
    float*    z    = (float*)   carve((size_t)NNODES * 512 * 4);   // projections
    float*    agg  = (float*)   carve((size_t)NNODES * 512 * 4);   // aggregation
    float*    ebuf = (float*)   carve((size_t)NEDGES * NHEADS * 4);// edge scores/ex
    float*    el_  = (float*)   carve((size_t)NNODES * NHEADS * 4);
    float*    er_  = (float*)   carve((size_t)NNODES * NHEADS * 4);
    unsigned* menc = (unsigned*)carve((size_t)NNODES * NHEADS * 4);
    float*    sden = (float*)   carve((size_t)NNODES * NHEADS * 4);
    float*    hbuf = (float*)   carve((size_t)NNODES * 64 * 4);    // inter-layer h
    float*    wt   = (float*)   carve((size_t)512 * 256 * 4);      // transposed+padded W

    const int Fdim[3] = { 64, 64, NCLS };
    const int Din [3] = { 256, 64, 64 };

    const float* hin = x;
    for (int L = 0; L < 3; ++L) {
        const int F     = Fdim[L];
        const int Dout  = NHEADS * F;
        const int K     = Din[L];
        const int Ndpad = (Dout + 31) & ~31;

        hipMemsetAsync(agg,  0, (size_t)NNODES * Dout * 4, stream);
        hipMemsetAsync(sden, 0, (size_t)NNODES * NHEADS * 4, stream);
        {
            int n = NNODES * NHEADS;
            gat_fill_u32<<<(n + 255) / 256, 256, 0, stream>>>(menc, ENC_NEG_INF, n);
        }

        // Wt = pad(W^T) : [Ndpad, K]
        {
            int n = Ndpad * K;
            gat_transpose_pad<<<(n + 255) / 256, 256, 0, stream>>>(W[L], wt, K, Dout, Ndpad);
        }

        // z = hin @ W  (WMMA bf16, f32 accum); 16x32 tile per wave
        {
            dim3 grid(NNODES / 16, Ndpad / 32);
            gat_gemm_wmma<<<grid, 32, 0, stream>>>(hin, wt, z, K, Dout);
        }

        // el/er
        {
            int n = NNODES * NHEADS;
            gat_attn_terms<<<(n + 255) / 256, 256, 0, stream>>>(z, al[L], ar[L], el_, er_, F);
        }

        // edge scores + segment max / exp / sum
        {
            int n = NEDGES * NHEADS;
            gat_edge_score_max<<<(n + 255) / 256, 256, 0, stream>>>(el_, er_, src, dst, ebuf, menc);
            gat_edge_exp_sum  <<<(n + 255) / 256, 256, 0, stream>>>(ebuf, menc, sden, dst);
        }

        // alpha-weighted scatter-aggregate
        gat_aggregate<<<NEDGES, 128, 0, stream>>>(z, ebuf, sden, src, dst, agg, Dout, F);

        if (L < 2) {
            int n = NNODES * F;
            gat_finalize_hidden<<<(n + 255) / 256, 256, 0, stream>>>(agg, bb[L], hbuf, F);
            hin = hbuf;
        } else {
            gat_finalize_softmax<<<(NNODES + 127) / 128, 128, 0, stream>>>(agg, bb[2], (float*)d_out);
        }
    }
}
